// freeSolveModel_86655260164544
// MI455X (gfx1250) — matrix operand order repaired and checked by reading, not compiled, not run on hardware
//
#include <hip/hip_runtime.h>

typedef __attribute__((ext_vector_type(16))) _Float16 v16h;
typedef __attribute__((ext_vector_type(8)))  float    v8f;

#define WPB 4  // waves per block in GEMM

// ---------- helpers ----------
__device__ inline unsigned enc_f(float f) {
  unsigned b = __float_as_uint(f);
  return (b & 0x80000000u) ? ~b : (b | 0x80000000u);
}
__device__ inline float dec_f(unsigned u) {
  unsigned b = (u & 0x80000000u) ? (u ^ 0x80000000u) : ~u;
  return __uint_as_float(b);
}
__device__ inline void edge_sd(const int* ei, int E, int e, int& s, int& d) {
  if (e < E) { s = ei[e]; d = ei[E + e]; } else { s = e - E; d = e - E; }
}

// ---------- trivial kernels ----------
__global__ void k_fill(float* p, int n, float v) {
  int i = blockIdx.x * blockDim.x + threadIdx.x;
  if (i < n) p[i] = v;
}
__global__ void k_edge_deg(const int* ei, int E, float* deg) {
  int e = blockIdx.x * blockDim.x + threadIdx.x;
  if (e < E) atomicAdd(&deg[ei[E + e]], 1.0f);
}
__global__ void k_node_cnt(const int* batch, int N, float* cnt) {
  int i = blockIdx.x * blockDim.x + threadIdx.x;
  if (i < N) atomicAdd(&cnt[batch[i]], 1.0f);
}
__global__ void k_dinv(float* deg, int N) {
  int i = blockIdx.x * blockDim.x + threadIdx.x;
  if (i < N) deg[i] = rsqrtf(fmaxf(deg[i], 1.0f));
}
__global__ void k_cntinv(float* cnt, int G) {
  int g = blockIdx.x * blockDim.x + threadIdx.x;
  if (g < G) cnt[g] = 1.0f / fmaxf(cnt[g], 1.0f);
}
__global__ void k_bias_tanh(float* h, const float* bias, int n, int F) {
  int i = blockIdx.x * blockDim.x + threadIdx.x;
  if (i < n) h[i] = tanhf(h[i] + bias[i % F]);
}
__global__ void k_bias_add(float* h, const float* bias, int n, int F) {
  int i = blockIdx.x * blockDim.x + threadIdx.x;
  if (i < n) h[i] += bias[i % F];
}

// ---------- WMMA GEMM:  C[N,Nout] = act(A[N,K] @ W[K,Nout] + bias) ----------
// A-frag (16x32 f16): lane&15 = row, K idx = kt*32 + (j&4?16:0) + half*8 + (j&3)*2 (+s)
// B-frag (32x16 f16): lane&15 = col, K idx = kt*32 + (j&4?16:0) + half*8 + (j&3)*2 (+s)
// C/D (16x16 f32): row = v + 8*half, col = lane&15
// K templated -> whole-K A fragment set stays resident in VGPRs across all
// output-column tiles. BIAS templated -> no per-element branch in epilogue.
template <int ACT, int K, bool BIAS>
__global__ void k_gemm_wmma(const float* __restrict__ A, const float* __restrict__ W,
                            const float* __restrict__ bias, float* __restrict__ C,
                            int N, int Nout) {
  __shared__ _Float16 sW[16384];  // W transposed: sW[n*K + k], K*Nout <= 16384
  int tid = threadIdx.x;
  int total = K * Nout;
  for (int idx = tid; idx < total; idx += blockDim.x) {
    int k = idx / Nout, n = idx % Nout;
    sW[n * K + k] = (_Float16)W[idx];
  }
  __syncthreads();

  int wave = tid >> 5, lane = tid & 31;
  int half = lane >> 4, r = lane & 15;
  long rowTile = ((long)blockIdx.x * WPB + wave) * 16;
  if (rowTile >= N) return;  // wave-uniform: EXEC stays all-ones for WMMA

  constexpr int KT = K / 32;
  const float* arow = A + (rowTile + r) * (long)K;

  v16h a[KT];
#pragma unroll
  for (int kt = 0; kt < KT; ++kt) {
#pragma unroll
    for (int j = 0; j < 8; ++j) {
      int ka = kt * 32 + ((j & 4) ? 16 : 0) + half * 8 + (j & 3) * 2;
      float2 f = *reinterpret_cast<const float2*>(arow + ka);
      a[kt][2 * j]     = (_Float16)f.x;
      a[kt][2 * j + 1] = (_Float16)f.y;
    }
  }

  int ntiles = Nout >> 4;
  for (int nt = 0; nt < ntiles; ++nt) {
    v8f acc = {};
    const _Float16* wcol = sW + (nt * 16 + r) * K;
#pragma unroll
    for (int kt = 0; kt < KT; ++kt) {
      v16h b;
#pragma unroll
      for (int j = 0; j < 8; ++j) {
        int kb = kt * 32 + ((j & 4) ? 16 : 0) + half * 8 + (j & 3) * 2;
        b[2 * j]     = wcol[kb];
        b[2 * j + 1] = wcol[kb + 1];
      }
      acc = __builtin_amdgcn_wmma_f32_16x16x32_f16(false, a[kt], false, b,
                                                   (short)0, acc, false, false);
    }
    int col = nt * 16 + r;
    float bv = BIAS ? bias[col] : 0.0f;   // invariant across the 8 rows
    float* cp = C + (rowTile + 8 * half) * (long)Nout + col;
#pragma unroll
    for (int v = 0; v < 8; ++v) {
      float val = acc[v] + bv;
      if (ACT == 1) val = tanhf(val);
      cp[(long)v * Nout] = val;
    }
  }
}

// ---------- GCN aggregation (F = 64) ----------
__global__ void k_gcn_self(const float* t, const float* dinv, float* agg, int N) {
  int i = blockIdx.x * blockDim.x + threadIdx.x;
  if (i < N * 64) {
    int n = i >> 6;
    float d = dinv[n];
    agg[i] = d * d * t[i];
  }
}
__global__ void k_gcn_scatter(const int* ei, int E, const float* __restrict__ t,
                              const float* __restrict__ dinv, float* agg) {
  int tid = blockIdx.x * blockDim.x + threadIdx.x;
  if (tid >= E * 16) return;
  int e = tid >> 4, q = tid & 15;
  int s = ei[e], d = ei[E + e];
  float w = dinv[s] * dinv[d];
  float4 v = *reinterpret_cast<const float4*>(t + (long)s * 64 + q * 4);
  float* o = agg + (long)d * 64 + q * 4;
  atomicAdd(o + 0, w * v.x); atomicAdd(o + 1, w * v.y);
  atomicAdd(o + 2, w * v.z); atomicAdd(o + 3, w * v.w);
}

// ---------- GATv2 edge passes ----------
__global__ void k_gat_logits(const int* ei, int E, int EA, int H,
                             const float* __restrict__ xl, const float* __restrict__ xr,
                             const float* __restrict__ att, float* logit, unsigned* maxu) {
  int tid = blockIdx.x * blockDim.x + threadIdx.x;
  if (tid >= EA * H) return;
  int h = tid % H, e = tid / H;
  int s, d; edge_sd(ei, E, e, s, d);
  int HC = H * 64;
  const float* pl = xl + (long)s * HC + h * 64;
  const float* pr = xr + (long)d * HC + h * 64;
  const float* pa = att + h * 64;
  float acc = 0.f;
#pragma unroll 8
  for (int c = 0; c < 64; ++c) {
    float v = pl[c] + pr[c];
    v = (v > 0.f) ? v : 0.2f * v;  // leaky_relu, slope 0.2
    acc += v * pa[c];
  }
  logit[tid] = acc;
  atomicMax(&maxu[(long)d * H + h], enc_f(acc));
}
__global__ void k_gat_p(const int* ei, int E, int EA, int H,
                        float* logit, const unsigned* maxu, float* denom) {
  int tid = blockIdx.x * blockDim.x + threadIdx.x;
  if (tid >= EA * H) return;
  int h = tid % H, e = tid / H;
  int s, d; edge_sd(ei, E, e, s, d);
  unsigned u = maxu[(long)d * H + h];
  float m = (u == 0u) ? 0.f : dec_f(u);
  float p = expf(logit[tid] - m);
  logit[tid] = p;
  atomicAdd(&denom[(long)d * H + h], p);
}
__global__ void k_gat_scatter(const int* ei, int E, int EA, int H,
                              const float* __restrict__ p, const float* __restrict__ denom,
                              const float* __restrict__ xl, float* gout) {
  int tid = blockIdx.x * blockDim.x + threadIdx.x;
  if (tid >= EA * H * 16) return;
  int q = tid & 15;
  int eh = tid >> 4;
  int h = eh % H, e = eh / H;
  int s, d; edge_sd(ei, E, e, s, d);
  float alpha = p[eh] / (denom[(long)d * H + h] + 1e-16f);
  int HC = H * 64;
  float4 v = *reinterpret_cast<const float4*>(xl + (long)s * HC + h * 64 + q * 4);
  float* o = gout + (long)d * HC + h * 64 + q * 4;
  atomicAdd(o + 0, alpha * v.x); atomicAdd(o + 1, alpha * v.y);
  atomicAdd(o + 2, alpha * v.z); atomicAdd(o + 3, alpha * v.w);
}

// ---------- GraphNorm ----------
__global__ void k_seg_sum(const float* x, const int* batch, float* gsum, int N) {
  int tid = blockIdx.x * blockDim.x + threadIdx.x;
  if (tid >= N * 64) return;
  int n = tid >> 6, f = tid & 63;
  atomicAdd(&gsum[(long)batch[n] * 64 + f], x[tid]);
}
__global__ void k_gn_xc(const float* x, const int* batch, const float* gsum,
                        const float* cntinv, const float* a, float* xc, float* gvar, int N) {
  int tid = blockIdx.x * blockDim.x + threadIdx.x;
  if (tid >= N * 64) return;
  int n = tid >> 6, f = tid & 63;
  int g = batch[n];
  float mean = gsum[(long)g * 64 + f] * cntinv[g];
  float v = x[tid] - a[f] * mean;
  xc[tid] = v;
  atomicAdd(&gvar[(long)g * 64 + f], v * v);
}
__global__ void k_gn_out(const float* xc, const int* batch, const float* gvar,
                         const float* cntinv, const float* w, const float* b,
                         float* out, int N) {
  int tid = blockIdx.x * blockDim.x + threadIdx.x;
  if (tid >= N * 64) return;
  int n = tid >> 6, f = tid & 63;
  int g = batch[n];
  float var = gvar[(long)g * 64 + f] * cntinv[g];
  out[tid] = w[f] * xc[tid] * rsqrtf(var + 2e-5f) + b[f];
}

// ---------- Readout ----------
__global__ void k_readout_acc(const float* h, const int* batch, unsigned* gmax,
                              float* gsum, int N) {
  int tid = blockIdx.x * blockDim.x + threadIdx.x;
  if (tid >= N * 64) return;
  int n = tid >> 6, f = tid & 63;
  int g = batch[n];
  float v = h[tid];
  atomicMax(&gmax[(long)g * 64 + f], enc_f(v));
  atomicAdd(&gsum[(long)g * 64 + f], v);
}
__global__ void k_hidden(const unsigned* gmax, const float* gsum, const float* cntinv,
                         float* hid, int G) {
  int tid = blockIdx.x * blockDim.x + threadIdx.x;
  if (tid >= G * 64) return;
  int g = tid >> 6, f = tid & 63;
  unsigned u = gmax[tid];
  hid[(long)g * 128 + f] = (u == 0u) ? 0.f : dec_f(u);
  hid[(long)g * 128 + 64 + f] = gsum[tid] * cntinv[g];
}
__global__ void k_outv(const float* hid, const float* Wout, const float* bout,
                       float* out, int G) {
  int g = blockIdx.x * blockDim.x + threadIdx.x;
  if (g >= G) return;
  float acc = bout[0];
#pragma unroll 8
  for (int k = 0; k < 128; ++k) acc += hid[(long)g * 128 + k] * Wout[k];
  out[g] = acc;
}

// ---------- launch ----------
static inline unsigned nblk(long n, int b) { return (unsigned)((n + b - 1) / b); }

extern "C" void kernel_launch(void* const* d_in, const int* in_sizes, int n_in,
                              void* d_out, int out_size, void* d_ws, size_t ws_size,
                              hipStream_t stream) {
  const float* x      = (const float*)d_in[0];
  const int*   ei     = (const int*)d_in[1];
  const int*   batch  = (const int*)d_in[2];
  const float* W_in   = (const float*)d_in[3];
  const float* b_in   = (const float*)d_in[4];
  const float* Wl1    = (const float*)d_in[5];
  const float* Wr1    = (const float*)d_in[6];
  const float* att1   = (const float*)d_in[7];
  const float* b1     = (const float*)d_in[8];
  const float* W_lin1 = (const float*)d_in[9];
  const float* b_lin1 = (const float*)d_in[10];
  const float* gn1_w  = (const float*)d_in[11];
  const float* gn1_b  = (const float*)d_in[12];
  const float* gn1_a  = (const float*)d_in[13];
  const float* W2     = (const float*)d_in[14];
  const float* b2     = (const float*)d_in[15];
  const float* gn2_w  = (const float*)d_in[16];
  const float* gn2_b  = (const float*)d_in[17];
  const float* gn2_a  = (const float*)d_in[18];
  const float* Wl3    = (const float*)d_in[19];
  const float* Wr3    = (const float*)d_in[20];
  const float* att3   = (const float*)d_in[21];
  const float* b3     = (const float*)d_in[22];
  const float* W_lin2 = (const float*)d_in[23];
  const float* b_lin2 = (const float*)d_in[24];
  const float* W4     = (const float*)d_in[25];
  const float* b4     = (const float*)d_in[26];
  const float* W_out  = (const float*)d_in[27];
  const float* b_out  = (const float*)d_in[28];

  const int N = in_sizes[0] / 64;
  const int E = in_sizes[1] / 2;
  const int G = out_size / 129;     // out(G) + hidden(G*128)
  const int EA = E + N;

  // workspace partition (floats)
  float* ws = (float*)d_ws;
  size_t o = 0;
  float* dinv   = ws + o; o += (size_t)N;
  float* cntinv = ws + o; o += (size_t)G;
  float* hA     = ws + o; o += (size_t)N * 64;
  float* hB     = ws + o; o += (size_t)N * 64;
  float* xl     = ws + o; o += (size_t)N * 256;
  float* xr     = ws + o; o += (size_t)N * 256;   // doubles as GAT output buffer
  float* logit  = ws + o; o += (size_t)EA * 4;
  unsigned* maxu = (unsigned*)(ws + o); o += (size_t)N * 4;
  float* denom  = ws + o; o += (size_t)N * 4;
  float* gA     = ws + o; o += (size_t)G * 64;
  float* gB     = ws + o; o += (size_t)G * 64;
  (void)ws_size; (void)n_in;

  const int B = 256;
  int gtiles = (N + 15) / 16;
  unsigned gemm_grid = nblk(gtiles, WPB);

  // ---- degrees / counts ----
  k_fill<<<nblk(N, B), B, 0, stream>>>(dinv, N, 1.0f);   // self-loop contributes 1
  k_fill<<<nblk(G, B), B, 0, stream>>>(cntinv, G, 0.0f);
  k_edge_deg<<<nblk(E, B), B, 0, stream>>>(ei, E, dinv);
  k_node_cnt<<<nblk(N, B), B, 0, stream>>>(batch, N, cntinv);
  k_dinv<<<nblk(N, B), B, 0, stream>>>(dinv, N);
  k_cntinv<<<nblk(G, B), B, 0, stream>>>(cntinv, G);

  // ---- GCN in: h = tanh(agg(x @ W_in) + b_in) -> hA ----
  k_gemm_wmma<0, 64, false><<<gemm_grid, 128, 0, stream>>>(x, W_in, nullptr, hB, N, 64);
  k_gcn_self<<<nblk((long)N * 64, B), B, 0, stream>>>(hB, dinv, hA, N);
  k_gcn_scatter<<<nblk((long)E * 16, B), B, 0, stream>>>(ei, E, hB, dinv, hA);
  k_bias_tanh<<<nblk((long)N * 64, B), B, 0, stream>>>(hA, b_in, N * 64, 64);

  // ---- GATv2 #1 (H=4) -> gout(=xr), then lin1+tanh -> hB ----
  k_gemm_wmma<0, 64, false><<<gemm_grid, 128, 0, stream>>>(hA, Wl1, nullptr, xl, N, 256);
  k_gemm_wmma<0, 64, false><<<gemm_grid, 128, 0, stream>>>(hA, Wr1, nullptr, xr, N, 256);
  k_fill<<<nblk((long)N * 4, B), B, 0, stream>>>((float*)maxu, N * 4, 0.0f);
  k_fill<<<nblk((long)N * 4, B), B, 0, stream>>>(denom, N * 4, 0.0f);
  k_gat_logits<<<nblk((long)EA * 4, B), B, 0, stream>>>(ei, E, EA, 4, xl, xr, att1, logit, maxu);
  k_gat_p<<<nblk((long)EA * 4, B), B, 0, stream>>>(ei, E, EA, 4, logit, maxu, denom);
  k_fill<<<nblk((long)N * 256, B), B, 0, stream>>>(xr, N * 256, 0.0f);  // xr now = gout
  k_gat_scatter<<<nblk((long)EA * 4 * 16, B), B, 0, stream>>>(ei, E, EA, 4, logit, denom, xl, xr);
  k_bias_add<<<nblk((long)N * 256, B), B, 0, stream>>>(xr, b1, N * 256, 256);
  k_gemm_wmma<1, 256, true><<<gemm_grid, 128, 0, stream>>>(xr, W_lin1, b_lin1, hB, N, 64);

  // ---- GraphNorm 1: hB -> hA (xc staged in xl) ----
  k_fill<<<nblk((long)G * 64, B), B, 0, stream>>>(gA, G * 64, 0.0f);
  k_fill<<<nblk((long)G * 64, B), B, 0, stream>>>(gB, G * 64, 0.0f);
  k_seg_sum<<<nblk((long)N * 64, B), B, 0, stream>>>(hB, batch, gA, N);
  k_gn_xc<<<nblk((long)N * 64, B), B, 0, stream>>>(hB, batch, gA, cntinv, gn1_a, xl, gB, N);
  k_gn_out<<<nblk((long)N * 64, B), B, 0, stream>>>(xl, batch, gB, cntinv, gn1_w, gn1_b, hA, N);

  // ---- GCN 2: hA -> hB ----
  k_gemm_wmma<0, 64, false><<<gemm_grid, 128, 0, stream>>>(hA, W2, nullptr, xr, N, 64);
  k_gcn_self<<<nblk((long)N * 64, B), B, 0, stream>>>(xr, dinv, hB, N);
  k_gcn_scatter<<<nblk((long)E * 16, B), B, 0, stream>>>(ei, E, xr, dinv, hB);
  k_bias_tanh<<<nblk((long)N * 64, B), B, 0, stream>>>(hB, b2, N * 64, 64);

  // ---- GraphNorm 2: hB -> hA ----
  k_fill<<<nblk((long)G * 64, B), B, 0, stream>>>(gA, G * 64, 0.0f);
  k_fill<<<nblk((long)G * 64, B), B, 0, stream>>>(gB, G * 64, 0.0f);
  k_seg_sum<<<nblk((long)N * 64, B), B, 0, stream>>>(hB, batch, gA, N);
  k_gn_xc<<<nblk((long)N * 64, B), B, 0, stream>>>(hB, batch, gA, cntinv, gn2_a, xl, gB, N);
  k_gn_out<<<nblk((long)N * 64, B), B, 0, stream>>>(xl, batch, gB, cntinv, gn2_w, gn2_b, hA, N);

  // ---- GATv2 #2 (H=2) -> gout(=xr), then lin2+tanh -> hB ----
  k_gemm_wmma<0, 64, false><<<gemm_grid, 128, 0, stream>>>(hA, Wl3, nullptr, xl, N, 128);
  k_gemm_wmma<0, 64, false><<<gemm_grid, 128, 0, stream>>>(hA, Wr3, nullptr, xr, N, 128);
  k_fill<<<nblk((long)N * 2, B), B, 0, stream>>>((float*)maxu, N * 2, 0.0f);
  k_fill<<<nblk((long)N * 2, B), B, 0, stream>>>(denom, N * 2, 0.0f);
  k_gat_logits<<<nblk((long)EA * 2, B), B, 0, stream>>>(ei, E, EA, 2, xl, xr, att3, logit, maxu);
  k_gat_p<<<nblk((long)EA * 2, B), B, 0, stream>>>(ei, E, EA, 2, logit, maxu, denom);
  k_fill<<<nblk((long)N * 128, B), B, 0, stream>>>(xr, N * 128, 0.0f);
  k_gat_scatter<<<nblk((long)EA * 2 * 16, B), B, 0, stream>>>(ei, E, EA, 2, logit, denom, xl, xr);
  k_bias_add<<<nblk((long)N * 128, B), B, 0, stream>>>(xr, b3, N * 128, 128);
  k_gemm_wmma<1, 128, true><<<gemm_grid, 128, 0, stream>>>(xr, W_lin2, b_lin2, hB, N, 64);

  // ---- GCN 3: hB -> hA ----
  k_gemm_wmma<0, 64, false><<<gemm_grid, 128, 0, stream>>>(hB, W4, nullptr, xl, N, 64);
  k_gcn_self<<<nblk((long)N * 64, B), B, 0, stream>>>(xl, dinv, hA, N);
  k_gcn_scatter<<<nblk((long)E * 16, B), B, 0, stream>>>(ei, E, xl, dinv, hA);
  k_bias_tanh<<<nblk((long)N * 64, B), B, 0, stream>>>(hA, b4, N * 64, 64);

  // ---- Readout: hidden -> d_out[G ...], out -> d_out[0..G) ----
  float* out_v  = (float*)d_out;
  float* hidden = (float*)d_out + G;
  k_fill<<<nblk((long)G * 64, B), B, 0, stream>>>(gA, G * 64, 0.0f);  // gmax (uint 0 == -inf)
  k_fill<<<nblk((long)G * 64, B), B, 0, stream>>>(gB, G * 64, 0.0f);  // gsum
  k_readout_acc<<<nblk((long)N * 64, B), B, 0, stream>>>(hA, batch, (unsigned*)gA, gB, N);
  k_hidden<<<nblk((long)G * 64, B), B, 0, stream>>>((unsigned*)gA, gB, cntinv, hidden, G);
  k_outv<<<nblk(G, B), B, 0, stream>>>(hidden, W_out, b_out, out_v, G);
}
